// VocalizationVAE_45853070852309
// MI455X (gfx1250) — matrix operand order repaired and checked by reading, not compiled, not run on hardware
//
#include <hip/hip_runtime.h>

// ---------------------------------------------------------------------------
// VocalizationVAE forward for MI455X (gfx1250, wave32, WMMA).
// GEMMs: v_wmma_f32_16x16x32_bf16, double-buffered LDS tiles staged with
// global_load_async_to_lds_b128 (ASYNCcnt), weights pre-transposed to [N,K].
// ---------------------------------------------------------------------------

typedef unsigned short ushort_t;
typedef __attribute__((ext_vector_type(4)))  unsigned int u32x4;
typedef __attribute__((ext_vector_type(16))) __bf16       v16bf;
typedef __attribute__((ext_vector_type(8)))  float        v8f;

__device__ __forceinline__ ushort_t f2bf(float f) {
  unsigned u = __float_as_uint(f);
  u += 0x7fffu + ((u >> 16) & 1u);  // round-to-nearest-even
  return (ushort_t)(u >> 16);
}
__device__ __forceinline__ float bf2f(ushort_t h) {
  return __uint_as_float(((unsigned)h) << 16);
}

union FragBF { v16bf v; u32x4 u[2]; };

// issue one 16B async global->LDS copy (per-lane addresses)
__device__ __forceinline__ void async_b128(const ushort_t* g, ushort_t* l) {
  unsigned lds = (unsigned)(unsigned long long)l;          // low 32b = LDS off
  unsigned long long ga = (unsigned long long)g;
  asm volatile("global_load_async_to_lds_b128 %0, %1, off"
               :: "v"(lds), "v"(ga) : "memory");
}
__device__ __forceinline__ void wait_async0() {
  asm volatile("s_wait_asynccnt 0x0" ::: "memory");
}

// ---------------------------------------------------------------------------
// Weight convert + transpose: src f32 [L][K][N] -> dst bf16 [L][N][K]
// 32x32 LDS tile, block 256 (32x8), grid (N/32, K/32, L)
// ---------------------------------------------------------------------------
__global__ __launch_bounds__(256) void k_cvt_T(const float* __restrict__ src,
                                               ushort_t* __restrict__ dst,
                                               int K, int N) {
  __shared__ float tile[32][33];
  const long lyr = blockIdx.z;
  const float*  s = src + lyr * (long)K * N;
  ushort_t*     d = dst + lyr * (long)K * N;
  const int n0 = blockIdx.x * 32, k0 = blockIdx.y * 32;
  const int tx = threadIdx.x & 31, ty = threadIdx.x >> 5;  // 32 x 8
  for (int r = ty; r < 32; r += 8)
    tile[r][tx] = s[(long)(k0 + r) * N + n0 + tx];
  __syncthreads();
  for (int r = ty; r < 32; r += 8)
    d[(long)(n0 + r) * K + k0 + tx] = f2bf(tile[tx][r]);
}

// ---------------------------------------------------------------------------
// WMMA GEMM:  C[M,N] = A[M,K] (bf16) @ W[K,N] + bias, opt ReLU.
// Wt is the transposed weight, bf16 [N,K] (K-contiguous like A).
// Block tile 64x128, 8 waves x (32x32), K-step 32, double-buffered async LDS.
// M%64==0, N%128==0, K%32==0 guaranteed by model dims.
// ---------------------------------------------------------------------------
template <bool RELU, bool OUT_BF16>
__global__ __launch_bounds__(256) void k_gemm_bf16(
    const ushort_t* __restrict__ A, const ushort_t* __restrict__ Wt,
    const float* __restrict__ bias, void* __restrict__ Cout,
    int M, int N, int Kd) {
  __shared__ __align__(16) ushort_t As[2][64 * 32];   // [m][k]
  __shared__ __align__(16) ushort_t Bs[2][128 * 32];  // [n][k]

  const int tid  = threadIdx.x;
  const int lane = tid & 31;
  const int wid  = tid >> 5;
  const int wm   = wid & 1;   // M direction (2)
  const int wn   = wid >> 1;  // N direction (4)
  const long m0  = (long)blockIdx.y * 64;
  const long n0  = (long)blockIdx.x * 128;

  v8f acc[2][2];
#pragma unroll
  for (int i = 0; i < 2; ++i)
#pragma unroll
    for (int j = 0; j < 2; ++j)
#pragma unroll
      for (int r = 0; r < 8; ++r) acc[i][j][r] = 0.0f;

  // staging coords: A tile = 256 x 16B, B tile = 512 x 16B (2 per thread)
  const int a_row = tid >> 2;        // 0..63
  const int a_col = (tid & 3) * 8;   // 0,8,16,24
  const int b_row = tid >> 1;        // 0..127
  const int b_col = (tid & 1) * 16;  // 0 or 16 (two b128s)
  const int lmod  = lane & 15;
  const int lhalf = lane >> 4;

  const ushort_t* Ab = A + (m0 + a_row) * (long)Kd + a_col;
  const ushort_t* Bb = Wt + (n0 + b_row) * (long)Kd + b_col;
  ushort_t* Al = &As[0][a_row * 32 + a_col];    // buf stride 64*32
  ushort_t* Bl = &Bs[0][b_row * 32 + b_col];    // buf stride 128*32

  const int nk = Kd >> 5;
  // prologue: stage tile 0 into buffer 0
  async_b128(Ab, Al);
  async_b128(Bb, Bl);
  async_b128(Bb + 8, Bl + 8);
  wait_async0();
  __syncthreads();

  int cur = 0;
  for (int ki = 0; ki < nk; ++ki) {
    // async prefetch of next K tile into the other buffer
    if (ki + 1 < nk) {
      const int kn = (ki + 1) << 5;
      async_b128(Ab + kn, Al + (cur ^ 1) * 64 * 32);
      async_b128(Bb + kn, Bl + (cur ^ 1) * 128 * 32);
      async_b128(Bb + kn + 8, Bl + (cur ^ 1) * 128 * 32 + 8);
    }

    // fragments per documented CDNA5 VGPR layouts
    const ushort_t* Asc = As[cur];
    const ushort_t* Bsc = Bs[cur];
    FragBF afrag[2], bfrag[2];
#pragma unroll
    for (int t = 0; t < 2; ++t) {
      int ar = (wm * 32 + t * 16 + lmod) * 32;
      afrag[t].u[0] = *(const u32x4*)(&Asc[ar + lhalf * 8]);
      afrag[t].u[1] = *(const u32x4*)(&Asc[ar + 16 + lhalf * 8]);
      int bc = (wn * 32 + t * 16 + lmod) * 32;
      bfrag[t].u[0] = *(const u32x4*)(&Bsc[bc + lhalf * 8]);
      bfrag[t].u[1] = *(const u32x4*)(&Bsc[bc + 16 + lhalf * 8]);
    }
#pragma unroll
    for (int i = 0; i < 2; ++i)
#pragma unroll
      for (int j = 0; j < 2; ++j)
        acc[i][j] = __builtin_amdgcn_wmma_f32_16x16x32_bf16(
            false, afrag[i].v, false, bfrag[j].v, (short)0, acc[i][j],
            false, false);

    wait_async0();    // next tile landed
    __syncthreads();  // all waves done reading cur, next buffer visible
    cur ^= 1;
  }

  // epilogue: elem r -> row r + 8*(lane/16), col = lane%16
#pragma unroll
  for (int i = 0; i < 2; ++i) {
#pragma unroll
    for (int j = 0; j < 2; ++j) {
      long col   = n0 + wn * 32 + j * 16 + lmod;
      float bv   = bias[col];
      long rbase = m0 + wm * 32 + i * 16 + lhalf * 8;
#pragma unroll
      for (int r = 0; r < 8; ++r) {
        float v = acc[i][j][r] + bv;
        if (RELU) v = fmaxf(v, 0.0f);
        if (OUT_BF16)
          ((ushort_t*)Cout)[(rbase + r) * (long)N + col] = f2bf(v);
        else
          ((float*)Cout)[(rbase + r) * (long)N + col] = v;
      }
    }
  }
}

// ---------------------------------------------------------------------------
// Encoder embedding: h = concat(CLS, x@W_data + b_data) + emb_in; pad rows 0.
// ---------------------------------------------------------------------------
__global__ void k_embed_enc(const float* __restrict__ x,
                            const float* __restrict__ Wd,
                            const float* __restrict__ bd,
                            const float* __restrict__ emb,
                            float* __restrict__ h, ushort_t* __restrict__ hbf) {
  long row = blockIdx.x;  // b*1032 + s
  int b = (int)(row / 1032), s = (int)(row % 1032);
  for (int c = threadIdx.x; c < 512; c += blockDim.x) {
    float v;
    if (s == 0) {
      v = emb[c];
    } else if (s <= 1024) {
      float a = bd[c];
      const float* xr = x + ((long)b * 1024 + (s - 1)) * 16;
#pragma unroll
      for (int j = 0; j < 16; ++j) a += xr[j] * Wd[j * 512 + c];
      v = a + emb[(long)s * 512 + c];
    } else {
      v = 0.0f;  // padded rows (masked in attention)
    }
    h[row * 512 + c]   = v;
    hbf[row * 512 + c] = f2bf(v);
  }
}

// ---------------------------------------------------------------------------
// Block-sparse attention. One block per (b, query-block), wave per head.
// qkv bf16 [B*Sp, 1536] (q|k|v). 2 global + 5 window + 3 random key blocks.
// ---------------------------------------------------------------------------
__global__ __launch_bounds__(256) void k_sparse_attn(
    const ushort_t* __restrict__ qkv, ushort_t* __restrict__ ao,
    const int* __restrict__ rnd, int Sreal, int nb, int Sp) {
  __shared__ __align__(16) ushort_t Qs[8][8][64];  // [head][p][d]
  __shared__ float Sc[8][8][80];                   // [head][p][kc]
  __shared__ int s_idx[10];
  __shared__ int s_val[10];

  const int blk  = blockIdx.x;
  const int b    = blk / nb;
  const int nbk  = blk % nb;
  const int tid  = threadIdx.x;
  const int lane = tid & 31;
  const int hd   = tid >> 5;  // wave == head

  if (tid < 10) {
    int kk = tid, bi;
    int val = 1;
    if (kk < 2) {
      bi = kk;                                   // global blocks
    } else if (kk < 7) {
      int w = nbk + kk - 4;                      // window -2..+2
      val = (w >= 0 && w < nb) ? 1 : 0;
      bi  = w < 0 ? 0 : (w >= nb ? nb - 1 : w);  // clip
    } else {
      bi = rnd[nbk * 3 + (kk - 7)];              // random blocks
    }
    s_idx[kk] = bi;
    s_val[kk] = val;
  }

  // stage Q tile: 8 tokens x 512 cols, 16 bf16 per thread
  {
    int p   = tid >> 5;
    int off = (tid & 31) * 16;
    long row = (long)(b * Sp + nbk * 8 + p);
    const ushort_t* src = qkv + row * 1536 + off;
    u32x4 q0 = *(const u32x4*)(src);
    u32x4 q1 = *(const u32x4*)(src + 8);
    int hh = off >> 6, dd = off & 63;
    *(u32x4*)(&Qs[hh][p][dd])     = q0;
    *(u32x4*)(&Qs[hh][p][dd + 8]) = q1;
  }
  __syncthreads();

  // scores
  for (int rep = 0; rep < 3; ++rep) {
    int kc = lane + rep * 32;
    if (kc >= 80) break;
    int bi = s_idx[kc >> 3];
    int t  = bi * 8 + (kc & 7);
    int valid = (s_val[kc >> 3] != 0) && (t < Sreal);
    const ushort_t* krow = qkv + (long)(b * Sp + t) * 1536 + 512 + hd * 64;
    float accp[8];
#pragma unroll
    for (int p = 0; p < 8; ++p) accp[p] = 0.0f;
    for (int dc = 0; dc < 64; dc += 8) {
      u32x4 kv = *(const u32x4*)(krow + dc);
      const ushort_t* kp = (const ushort_t*)&kv;
      float kf[8];
#pragma unroll
      for (int i2 = 0; i2 < 8; ++i2) kf[i2] = bf2f(kp[i2]);
#pragma unroll
      for (int p = 0; p < 8; ++p) {
        float s = 0.0f;
#pragma unroll
        for (int i2 = 0; i2 < 8; ++i2) s += bf2f(Qs[hd][p][dc + i2]) * kf[i2];
        accp[p] += s;
      }
    }
#pragma unroll
    for (int p = 0; p < 8; ++p)
      Sc[hd][p][kc] = valid ? accp[p] * 0.125f : -1.0e9f;
  }
  __syncthreads();

  // softmax per (head, row)
  for (int p = 0; p < 8; ++p) {
    float m = -1.0e30f;
    for (int kc = lane; kc < 80; kc += 32) m = fmaxf(m, Sc[hd][p][kc]);
#pragma unroll
    for (int o = 16; o > 0; o >>= 1) m = fmaxf(m, __shfl_xor(m, o, 32));
    float s = 0.0f;
    for (int kc = lane; kc < 80; kc += 32) {
      float e = __expf(Sc[hd][p][kc] - m);
      Sc[hd][p][kc] = e;
      s += e;
    }
#pragma unroll
    for (int o = 16; o > 0; o >>= 1) s += __shfl_xor(s, o, 32);
    float inv = 1.0f / s;
    for (int kc = lane; kc < 80; kc += 32) Sc[hd][p][kc] *= inv;
  }

  // out = attn @ V, each lane owns 2 head-dims
  {
    int d0 = lane * 2;
    float acc0[8], acc1[8];
#pragma unroll
    for (int p = 0; p < 8; ++p) { acc0[p] = 0.0f; acc1[p] = 0.0f; }
    for (int kc = 0; kc < 80; ++kc) {
      int bi = s_idx[kc >> 3];
      int t  = bi * 8 + (kc & 7);
      const ushort_t* vrow =
          qkv + (long)(b * Sp + t) * 1536 + 1024 + hd * 64 + d0;
      unsigned vv = *(const unsigned*)vrow;
      float v0 = bf2f((ushort_t)(vv & 0xffffu));
      float v1 = bf2f((ushort_t)(vv >> 16));
#pragma unroll
      for (int p = 0; p < 8; ++p) {
        float a = Sc[hd][p][kc];
        acc0[p] += a * v0;
        acc1[p] += a * v1;
      }
    }
#pragma unroll
    for (int p = 0; p < 8; ++p) {
      long row = (long)(b * Sp + nbk * 8 + p);
      unsigned pk = (unsigned)f2bf(acc0[p]) | ((unsigned)f2bf(acc1[p]) << 16);
      *(unsigned*)(ao + row * 512 + hd * 64 + d0) = pk;
    }
  }
}

// ---------------------------------------------------------------------------
// y = LayerNorm(x + r)*g + b   (row 512, one row/block); writes f32 + bf16
// ---------------------------------------------------------------------------
__global__ __launch_bounds__(256) void k_add_ln(
    const float* __restrict__ x, const float* __restrict__ r,
    const float* __restrict__ g, const float* __restrict__ bta,
    float* __restrict__ y, ushort_t* __restrict__ ybf) {
  __shared__ float red1[256];
  __shared__ float red2[256];
  long row = blockIdx.x;
  int tid  = threadIdx.x;
  const float* xr = x + row * 512;
  const float* rr = r + row * 512;
  float v0 = xr[2 * tid]     + rr[2 * tid];
  float v1 = xr[2 * tid + 1] + rr[2 * tid + 1];
  red1[tid] = v0 + v1;
  red2[tid] = v0 * v0 + v1 * v1;
  __syncthreads();
  for (int s = 128; s > 0; s >>= 1) {
    if (tid < s) {
      red1[tid] += red1[tid + s];
      red2[tid] += red2[tid + s];
    }
    __syncthreads();
  }
  float mean = red1[0] * (1.0f / 512.0f);
  float var  = red2[0] * (1.0f / 512.0f) - mean * mean;
  float rstd = rsqrtf(var + 1e-5f);
  float o0 = (v0 - mean) * rstd * g[2 * tid]     + bta[2 * tid];
  float o1 = (v1 - mean) * rstd * g[2 * tid + 1] + bta[2 * tid + 1];
  y[row * 512 + 2 * tid]       = o0;
  y[row * 512 + 2 * tid + 1]   = o1;
  ybf[row * 512 + 2 * tid]     = f2bf(o0);
  ybf[row * 512 + 2 * tid + 1] = f2bf(o1);
}

// ---------------------------------------------------------------------------
// Latent from pooled CLS row; z = mean + eps*exp(0.5*lv); ELBO partial per b.
// ---------------------------------------------------------------------------
__global__ void k_latent(const float* __restrict__ h,
                         const float* __restrict__ eps,
                         const float* __restrict__ Wm,
                         const float* __restrict__ bm,
                         const float* __restrict__ Wl,
                         const float* __restrict__ bl,
                         float* __restrict__ z, float* __restrict__ elbo_b,
                         int Sp) {
  int b = blockIdx.x, lane = threadIdx.x;
  const float* pooled = h + (long)b * Sp * 512;
  float e = 0.0f;
  if (lane < 16) {
    float m = bm[lane], lv = bl[lane];
    for (int d = 0; d < 512; ++d) {
      float p = pooled[d];
      m  += p * Wm[d * 16 + lane];
      lv += p * Wl[d * 16 + lane];
    }
    z[b * 16 + lane] = m + eps[b * 16 + lane] * __expf(0.5f * lv);
    e = 1.0f + lv - m * m - __expf(lv);
  }
#pragma unroll
  for (int o = 16; o > 0; o >>= 1) e += __shfl_xor(e, o, 32);
  if (lane == 0) elbo_b[b] = -0.5f * e;
}

__global__ void k_seq(const float* __restrict__ z, const float* __restrict__ We,
                      const float* __restrict__ be, float* __restrict__ seq) {
  int gid = blockIdx.x * blockDim.x + threadIdx.x;
  int b = gid >> 11, j = gid & 2047;
  float a = be[j];
#pragma unroll
  for (int l = 0; l < 16; ++l) a += z[b * 16 + l] * We[l * 2048 + j];
  seq[gid] = a;
}

__global__ void k_dec_embed(const float* __restrict__ seq,
                            const float* __restrict__ Wc,
                            const float* __restrict__ bc,
                            const float* __restrict__ embo,
                            float* __restrict__ h,
                            ushort_t* __restrict__ hbf) {
  long row = blockIdx.x;  // b*1024 + s
  int b = (int)(row >> 10), s = (int)(row & 1023);
  float s0 = seq[b * 2048 + s];
  float s1 = seq[b * 2048 + 1024 + s];
  for (int c = threadIdx.x; c < 512; c += blockDim.x) {
    float v = s0 * Wc[c] + s1 * Wc[512 + c] + bc[c] + embo[(long)s * 512 + c];
    h[row * 512 + c]   = v;
    hbf[row * 512 + c] = f2bf(v);
  }
}

__global__ __launch_bounds__(256) void k_head(const float* __restrict__ h,
                                              const float* __restrict__ Ws,
                                              const float* __restrict__ bs,
                                              float* __restrict__ out) {
  int t = blockIdx.x * 256 + threadIdx.x;  // 32768*16 total
  int row = t >> 4, j = t & 15;
  const float* hr = h + (long)row * 512;
  float a = bs[j];
  for (int d = 0; d < 512; ++d) a += hr[d] * Ws[d * 16 + j];
  out[t] = a;
}

__global__ void k_elbo(const float* __restrict__ elbo_b,
                       float* __restrict__ out) {
  int lane = threadIdx.x;
  float e = elbo_b[lane];
#pragma unroll
  for (int o = 16; o > 0; o >>= 1) e += __shfl_xor(e, o, 32);
  if (lane == 0) out[0] = e * (1.0f / 32.0f);
}

// ---------------------------------------------------------------------------
// Host driver
// ---------------------------------------------------------------------------
extern "C" void kernel_launch(void* const* d_in, const int* in_sizes, int n_in,
                              void* d_out, int out_size, void* d_ws,
                              size_t ws_size, hipStream_t stream) {
  (void)in_sizes; (void)n_in; (void)out_size; (void)ws_size;

  const float* x       = (const float*)d_in[0];
  const float* eps     = (const float*)d_in[1];
  const float* emb_in  = (const float*)d_in[2];
  const float* emb_out = (const float*)d_in[3];
  const float* W_data  = (const float*)d_in[4];
  const float* b_data  = (const float*)d_in[5];
  const float* Wqkv    = (const float*)d_in[6];
  const float* bqkv    = (const float*)d_in[7];
  const float* Wo      = (const float*)d_in[8];
  const float* bo      = (const float*)d_in[9];
  const float* ln1_g   = (const float*)d_in[10];
  const float* ln1_b   = (const float*)d_in[11];
  const float* W1      = (const float*)d_in[12];
  const float* b1      = (const float*)d_in[13];
  const float* W2      = (const float*)d_in[14];
  const float* b2      = (const float*)d_in[15];
  const float* ln2_g   = (const float*)d_in[16];
  const float* ln2_b   = (const float*)d_in[17];
  const float* W_mean  = (const float*)d_in[18];
  const float* b_mean  = (const float*)d_in[19];
  const float* W_lv    = (const float*)d_in[20];
  const float* b_lv    = (const float*)d_in[21];
  const float* W_exp   = (const float*)d_in[22];
  const float* b_exp   = (const float*)d_in[23];
  const float* W_conv  = (const float*)d_in[24];
  const float* b_conv  = (const float*)d_in[25];
  const float* W_seqp  = (const float*)d_in[26];
  const float* b_seqp  = (const float*)d_in[27];
  const int* rand_enc  = (const int*)d_in[28];
  const int* rand_dec  = (const int*)d_in[29];
  float* out = (float*)d_out;

  // ---- workspace carve-out ----
  char* ws = (char*)d_ws;
  size_t off = 0;
  auto carve = [&](size_t bytes) -> char* {
    char* p = ws + off;
    off += (bytes + 255) & ~(size_t)255;
    return p;
  };
  const long Me = 32 * 1032;  // encoder padded rows (multiple of 64)
  const long Md = 32 * 1024;  // decoder rows

  ushort_t* wqkv_bf = (ushort_t*)carve(10l * 512 * 1536 * 2);  // [L][N][K]
  ushort_t* wo_bf   = (ushort_t*)carve(10l * 512 * 512 * 2);
  ushort_t* w1_bf   = (ushort_t*)carve(10l * 512 * 2048 * 2);
  ushort_t* w2_bf   = (ushort_t*)carve(10l * 2048 * 512 * 2);
  float*    hf      = (float*)carve((size_t)Me * 512 * 4);
  ushort_t* hbf     = (ushort_t*)carve((size_t)Me * 512 * 2);
  ushort_t* qkv_bf  = (ushort_t*)carve((size_t)Me * 1536 * 2);
  ushort_t* ao_bf   = (ushort_t*)carve((size_t)Me * 512 * 2);
  float*    tmpf    = (float*)carve((size_t)Me * 512 * 4);
  ushort_t* ff_bf   = (ushort_t*)carve((size_t)Me * 2048 * 2);
  float*    z_ws    = (float*)carve(32 * 16 * 4);
  float*    elbo_b  = (float*)carve(32 * 4);
  float*    seq_ws  = (float*)carve(32 * 2048 * 4);

  // ---- weights -> bf16, transposed to [N,K] (K-contiguous for WMMA B) ----
  k_cvt_T<<<dim3(1536 / 32, 512 / 32, 10), 256, 0, stream>>>(Wqkv, wqkv_bf,
                                                             512, 1536);
  k_cvt_T<<<dim3(512 / 32, 512 / 32, 10), 256, 0, stream>>>(Wo, wo_bf, 512,
                                                            512);
  k_cvt_T<<<dim3(2048 / 32, 512 / 32, 10), 256, 0, stream>>>(W1, w1_bf, 512,
                                                             2048);
  k_cvt_T<<<dim3(512 / 32, 2048 / 32, 10), 256, 0, stream>>>(W2, w2_bf, 2048,
                                                             512);

  // ---- encoder embedding ----
  k_embed_enc<<<dim3((unsigned)Me), 128, 0, stream>>>(x, W_data, b_data, emb_in,
                                                      hf, hbf);

  // ---- encoder layers ----
  for (int i = 0; i < 5; ++i) {
    k_gemm_bf16<false, true><<<dim3(1536 / 128, (unsigned)(Me / 64)), 256, 0,
                               stream>>>(hbf, wqkv_bf + (size_t)i * 512 * 1536,
                                         bqkv + i * 1536, qkv_bf, (int)Me, 1536,
                                         512);
    k_sparse_attn<<<dim3(32 * 129), 256, 0, stream>>>(qkv_bf, ao_bf, rand_enc,
                                                      1025, 129, 1032);
    k_gemm_bf16<false, false><<<dim3(512 / 128, (unsigned)(Me / 64)), 256, 0,
                                stream>>>(ao_bf, wo_bf + (size_t)i * 512 * 512,
                                          bo + i * 512, tmpf, (int)Me, 512,
                                          512);
    k_add_ln<<<dim3((unsigned)Me), 256, 0, stream>>>(hf, tmpf, ln1_g + i * 512,
                                                     ln1_b + i * 512, hf, hbf);
    k_gemm_bf16<true, true><<<dim3(2048 / 128, (unsigned)(Me / 64)), 256, 0,
                              stream>>>(hbf, w1_bf + (size_t)i * 512 * 2048,
                                        b1 + i * 2048, ff_bf, (int)Me, 2048,
                                        512);
    k_gemm_bf16<false, false><<<dim3(512 / 128, (unsigned)(Me / 64)), 256, 0,
                                stream>>>(ff_bf, w2_bf + (size_t)i * 2048 * 512,
                                          b2 + i * 512, tmpf, (int)Me, 512,
                                          2048);
    k_add_ln<<<dim3((unsigned)Me), 256, 0, stream>>>(hf, tmpf, ln2_g + i * 512,
                                                     ln2_b + i * 512, hf, hbf);
  }

  // ---- latent + decoder seed ----
  k_latent<<<dim3(32), 32, 0, stream>>>(hf, eps, W_mean, b_mean, W_lv, b_lv,
                                        z_ws, elbo_b, 1032);
  k_seq<<<dim3((32 * 2048) / 256), 256, 0, stream>>>(z_ws, W_exp, b_exp,
                                                     seq_ws);
  k_dec_embed<<<dim3((unsigned)Md), 128, 0, stream>>>(seq_ws, W_conv, b_conv,
                                                      emb_out, hf, hbf);

  // ---- decoder layers ----
  for (int i = 5; i < 10; ++i) {
    k_gemm_bf16<false, true><<<dim3(1536 / 128, (unsigned)(Md / 64)), 256, 0,
                               stream>>>(hbf, wqkv_bf + (size_t)i * 512 * 1536,
                                         bqkv + i * 1536, qkv_bf, (int)Md, 1536,
                                         512);
    k_sparse_attn<<<dim3(32 * 128), 256, 0, stream>>>(qkv_bf, ao_bf, rand_dec,
                                                      1024, 128, 1024);
    k_gemm_bf16<false, false><<<dim3(512 / 128, (unsigned)(Md / 64)), 256, 0,
                                stream>>>(ao_bf, wo_bf + (size_t)i * 512 * 512,
                                          bo + i * 512, tmpf, (int)Md, 512,
                                          512);
    k_add_ln<<<dim3((unsigned)Md), 256, 0, stream>>>(hf, tmpf, ln1_g + i * 512,
                                                     ln1_b + i * 512, hf, hbf);
    k_gemm_bf16<true, true><<<dim3(2048 / 128, (unsigned)(Md / 64)), 256, 0,
                              stream>>>(hbf, w1_bf + (size_t)i * 512 * 2048,
                                        b1 + i * 2048, ff_bf, (int)Md, 2048,
                                        512);
    k_gemm_bf16<false, false><<<dim3(512 / 128, (unsigned)(Md / 64)), 256, 0,
                                stream>>>(ff_bf, w2_bf + (size_t)i * 2048 * 512,
                                          b2 + i * 512, tmpf, (int)Md, 512,
                                          2048);
    k_add_ln<<<dim3((unsigned)Md), 256, 0, stream>>>(hf, tmpf, ln2_g + i * 512,
                                                     ln2_b + i * 512, hf, hbf);
  }

  // ---- head + elbo ----
  k_head<<<dim3((unsigned)((Md * 16) / 256)), 256, 0, stream>>>(hf, W_seqp,
                                                                b_seqp, out);
  k_elbo<<<dim3(1), 32, 0, stream>>>(elbo_b, out + 32l * 1024 * 16);
}
